// SwinTransformerBlock_73512660238367
// MI455X (gfx1250) — matrix operand order repaired and checked by reading, not compiled, not run on hardware
//
#include <hip/hip_runtime.h>

typedef __attribute__((ext_vector_type(16))) __bf16 v16bf;
typedef __attribute__((ext_vector_type(8)))  float  v8f;
typedef __attribute__((ext_vector_type(4)))  unsigned v4u;
typedef __attribute__((ext_vector_type(4)))  int   v4i;
typedef __attribute__((ext_vector_type(8)))  int   v8i;

#define WMMA_BF16(a, b, c) \
  __builtin_amdgcn_wmma_f32_16x16x32_bf16(false, (a), false, (b), (short)0, (c), false, false)

#if __has_builtin(__builtin_amdgcn_global_load_async_to_lds_b128)
#define HAVE_ASYNC 1
#else
#define HAVE_ASYNC 0
#endif
#if __has_builtin(__builtin_amdgcn_tensor_load_to_lds)
#define HAVE_TDM 1
#if __has_include(<hip/amd_detail/amd_gfx1250_TDM.h>)
#define HAVE_TDM6 1
#else
#define HAVE_TDM6 0
#endif
#else
#define HAVE_TDM 0
#endif

typedef __attribute__((address_space(1))) v4i v4i_as1;
typedef __attribute__((address_space(3))) v4i v4i_as3;

// ---------------------------------------------------------------------------
// helpers
// ---------------------------------------------------------------------------
__device__ __forceinline__ void async_b128(const void* g, void* l) {
#if HAVE_ASYNC
  __builtin_amdgcn_global_load_async_to_lds_b128((v4i_as1*)g, (v4i_as3*)l, 0, 0);
#else
  (void)g; (void)l;
#endif
}

__device__ __forceinline__ void wait_async0() {
#if HAVE_ASYNC
#if __has_builtin(__builtin_amdgcn_s_wait_asynccnt)
  __builtin_amdgcn_s_wait_asynccnt(0);
#else
  asm volatile("s_wait_asynccnt 0x0" ::: "memory");
#endif
#endif
}

__device__ __forceinline__ void wait_tensor0() {
#if HAVE_TDM
#if __has_builtin(__builtin_amdgcn_s_wait_tensorcnt)
  __builtin_amdgcn_s_wait_tensorcnt(0);
#else
  asm volatile("s_wait_tensorcnt 0x0" ::: "memory");
#endif
#endif
}

__device__ __forceinline__ unsigned lds_off(const void* p) {
  return (unsigned)(unsigned long long)(const __attribute__((address_space(3))) void*)p;
}

// two contiguous 16B LDS chunks -> one 16-elem bf16 fragment
__device__ __forceinline__ v16bf frag_ld(const __bf16* p0, const __bf16* p1) {
  union { uint4 u[2]; v16bf v; } t;
  t.u[0] = *(const uint4*)p0;
  t.u[1] = *(const uint4*)p1;
  return t.v;
}

__device__ __forceinline__ float blk_sum256(float v, float* red) {
  #pragma unroll
  for (int o = 16; o > 0; o >>= 1) v += __shfl_xor(v, o, 32);
  int w = threadIdx.x >> 5;
  if ((threadIdx.x & 31) == 0) red[w] = v;
  __syncthreads();
  float t = red[0];
  #pragma unroll
  for (int i = 1; i < 8; ++i) t += red[i];
  __syncthreads();
  return t;
}

// ---------------------------------------------------------------------------
// Weight convert: f32 (N,K) row-major -> bf16 (N,K) row-major
// ---------------------------------------------------------------------------
__global__ void wconv_kernel(const float* __restrict__ W, __bf16* __restrict__ Wt,
                             size_t n) {
  size_t i = (size_t)blockIdx.x * 256 + threadIdx.x;
  if (i < n) Wt[i] = (__bf16)W[i];
}

// ---------------------------------------------------------------------------
// prep: LN1 + roll(-3,-3) + window partition -> bf16 X rows (512*147, 256)
// ---------------------------------------------------------------------------
__global__ void __launch_bounds__(256)
prep_kernel(const float* __restrict__ rgb, const float* __restrict__ hs,
            const float* __restrict__ dem, const float* __restrict__ g,
            const float* __restrict__ bvec, __bf16* __restrict__ X) {
  __shared__ float red[8];
  int r = blockIdx.x;
  int c = threadIdx.x;
  int b_ = r / 147, tt = r % 147;
  int m = tt / 49, t = tt % 49;
  int b = b_ >> 6, win = b_ & 63;
  int wy = win >> 3, wx = win & 7;
  int wh = t / 7, ww = t % 7;
  int h = (wy * 7 + wh + 3) % 56;
  int w = (wx * 7 + ww + 3) % 56;
  const float* src = (m == 0 ? rgb : (m == 1 ? hs : dem)) +
                     ((size_t)b * 3136 + h * 56 + w) * 256;
  float v = src[c];
  float mean = blk_sum256(v, red) * (1.0f / 256.0f);
  float d = v - mean;
  float var = blk_sum256(d * d, red) * (1.0f / 256.0f);
  float xh = d * rsqrtf(var + 1e-5f) * g[c] + bvec[c];
  X[(size_t)r * 256 + c] = (__bf16)xh;
}

// ---------------------------------------------------------------------------
// tile stage: A 128x32 (row-major (M,K)), B 64 cols x 32 k (weights (N,K))
// ---------------------------------------------------------------------------
__device__ __forceinline__ void gemm_issue(const __bf16* __restrict__ A,
                                           const __bf16* __restrict__ Bw,
                                           __bf16 (*As)[32], __bf16 (*Bs)[40],
                                           int bm, int bn, int K, int k0, int tid) {
  int rr = tid >> 1, cseg = (tid & 1) * 16;
  const __bf16* ga = A + (size_t)(bm + rr) * K + k0 + cseg;
  int nr = tid >> 2, seg = (tid & 3) * 8;
  const __bf16* gb = Bw + (size_t)(bn + nr) * K + k0 + seg;
#if HAVE_ASYNC
  async_b128(ga, &As[rr][cseg]);
  async_b128(ga + 8, &As[rr][cseg + 8]);
  async_b128(gb, &Bs[nr][seg]);
#else
  *(uint4*)&As[rr][cseg]     = *(const uint4*)ga;
  *(uint4*)&As[rr][cseg + 8] = *(const uint4*)(ga + 8);
  *(uint4*)&Bs[nr][seg]      = *(const uint4*)gb;
#endif
  __builtin_prefetch(ga + 64, 0, 0);       // K+2 tile ahead -> global_prefetch_b8
}

// ---------------------------------------------------------------------------
// Generic bf16 GEMM: C(M,N) = A(M,K) @ Bw(N,K)^T + bias, double-buffered LDS.
// Block tile 128x64, 8 waves (4x2), wave = 32x32 = 2x2 WMMA 16x16x32 frags.
// EPI: 0 = bf16 out, 1 = bf16 + exact GELU, 2 = f32 out, 3 = f32 + resid
// ---------------------------------------------------------------------------
template <int EPI>
__global__ void __launch_bounds__(256)
gemm_kernel(const __bf16* __restrict__ A, const __bf16* __restrict__ Bw,
            const float* __restrict__ bias, void* __restrict__ Cout,
            const float* __restrict__ resid, int M, int N, int K) {
  __shared__ __bf16 As[2][128][32];
  __shared__ __bf16 Bs[2][64][40];       // [n][k], row stride 80B (16B aligned)
  (void)M;
  int tid = threadIdx.x;
  int lane = tid & 31, wave = tid >> 5;
  int bm = blockIdx.x * 128;
  int bn = blockIdx.y * 64;
  int wm = (wave >> 1) * 32, wn = (wave & 1) * 32;
  int half = lane >> 4, l16 = lane & 15;

  v8f acc[2][2] = {};
  int nk = K >> 5;

  gemm_issue(A, Bw, As[0], Bs[0], bm, bn, K, 0, tid);
  for (int kt = 0; kt < nk; ++kt) {
    int cur = kt & 1;
    wait_async0();
    __syncthreads();
    if (kt + 1 < nk)
      gemm_issue(A, Bw, As[cur ^ 1], Bs[cur ^ 1], bm, bn, K, (kt + 1) * 32, tid);

    v16bf a[2], b[2];
    #pragma unroll
    for (int i = 0; i < 2; ++i) {
      const __bf16* Ar = &As[cur][wm + i * 16 + l16][0];
      a[i] = frag_ld(Ar + half * 8, Ar + 16 + half * 8);
      const __bf16* Br = &Bs[cur][wn + i * 16 + l16][0];
      b[i] = frag_ld(Br + half * 8, Br + 16 + half * 8);
    }
    #pragma unroll
    for (int i = 0; i < 2; ++i)
      #pragma unroll
      for (int j = 0; j < 2; ++j)
        acc[i][j] = WMMA_BF16(a[i], b[j], acc[i][j]);
  }

  #pragma unroll
  for (int i = 0; i < 2; ++i)
    #pragma unroll
    for (int j = 0; j < 2; ++j) {
      int col = bn + wn + j * 16 + l16;
      float bv = bias ? bias[col] : 0.0f;
      #pragma unroll
      for (int rr = 0; rr < 8; ++rr) {
        int row = bm + wm + i * 16 + half * 8 + rr;
        float v = acc[i][j][rr] + bv;
        size_t idx = (size_t)row * N + col;
        if (EPI == 0) {
          ((__bf16*)Cout)[idx] = (__bf16)v;
        } else if (EPI == 1) {
          float ge = 0.5f * v * (1.0f + erff(v * 0.70710678118654752f));
          ((__bf16*)Cout)[idx] = (__bf16)ge;
        } else if (EPI == 2) {
          ((float*)Cout)[idx] = v;
        } else {
          ((float*)Cout)[idx] = v + resid[idx];
        }
      }
    }
}

// ---------------------------------------------------------------------------
// Attention: one block per (window b_, head, 16-row m-tile).
// K/V staged by TDM (strided 2-D tile, OOB rows auto-zero) when available.
// ---------------------------------------------------------------------------
__global__ void __launch_bounds__(256)
attn_kernel(const __bf16* __restrict__ QKV, const float* __restrict__ rpb,
            const float* __restrict__ maskw, __bf16* __restrict__ AO) {
  __shared__ __bf16 sk[160][32];
  __shared__ __bf16 sv[160][32];
  __shared__ __bf16 sq[16][32];
  __shared__ float  ss[16][160];
  __shared__ __bf16 sp[16][168];         // row stride 336B (16B aligned)
  __shared__ float  sred[16][16];
  int b_ = blockIdx.x, hh = blockIdx.y, mt = blockIdx.z;
  int tid = threadIdx.x, lane = tid & 31, wave = tid >> 5;
  int half = lane >> 4, l16 = lane & 15;
  size_t base = (size_t)b_ * 147 * 768;

#if HAVE_TDM
  if (wave == 0) {
    unsigned long long gk = (unsigned long long)(uintptr_t)(QKV + base + 256 + hh * 32);
    unsigned long long gv = (unsigned long long)(uintptr_t)(QKV + base + 512 + hh * 32);
    // D# group1: data_size=2B, tensor_dim0=32, tensor_dim1=147,
    // tile 32x160 (rows 147..159 read OOB -> zero), stride0=768 elements.
    v8i g1 = {0x00010000, 32 << 16, 147 << 16, 32 << 16, 160, 768, 0, 0};
    v4i z4 = {0, 0, 0, 0};
    v4u g0k = {1u, lds_off(&sk[0][0]), (unsigned)gk,
               (unsigned)((gk >> 32) & 0x1FFFFFFu) | 0x80000000u};  // type=2
    v4u g0v = {1u, lds_off(&sv[0][0]), (unsigned)gv,
               (unsigned)((gv >> 32) & 0x1FFFFFFu) | 0x80000000u};
#if HAVE_TDM6
    v8i z8 = {0, 0, 0, 0, 0, 0, 0, 0};
    __builtin_amdgcn_tensor_load_to_lds(g0k, g1, z4, z4, z8, 0);
    __builtin_amdgcn_tensor_load_to_lds(g0v, g1, z4, z4, z8, 0);
#else
    __builtin_amdgcn_tensor_load_to_lds(g0k, g1, z4, z4, 0);
    __builtin_amdgcn_tensor_load_to_lds(g0v, g1, z4, z4, 0);
#endif
  }
  wait_tensor0();
#else
  for (int idx = tid; idx < 160 * 32; idx += 256) {
    int j = idx >> 5, d = idx & 31;
    __bf16 kv = (__bf16)0.0f, vv = (__bf16)0.0f;
    if (j < 147) {
      size_t o = base + (size_t)j * 768 + hh * 32 + d;
      kv = QKV[o + 256];
      vv = QKV[o + 512];
    }
    sk[j][d] = kv;  sv[j][d] = vv;
  }
#endif
  for (int idx = tid; idx < 16 * 32; idx += 256) {
    int rr = idx >> 5, d = idx & 31;
    int nq = mt * 16 + rr;
    sq[rr][d] = (nq < 147) ? QKV[base + (size_t)nq * 768 + hh * 32 + d] : (__bf16)0.0f;
  }
  __syncthreads();

  const __bf16* Aq = &sq[l16][0];
  v16bf aq = frag_ld(Aq + half * 8, Aq + 16 + half * 8);
  const float scale = 0.17677669529663687f;   // 32^-0.5
  int win = b_ & 63;

  // hoisted row-side (per accumulator row) index math
  int a49r[8], ahr[8], awr[8];
  #pragma unroll
  for (int rr = 0; rr < 8; ++rr) {
    int nq = mt * 16 + half * 8 + rr;  if (nq > 146) nq = 146;
    int a49 = nq % 49;
    a49r[rr] = a49;  ahr[rr] = a49 / 7;  awr[rr] = a49 % 7;
  }
  const float* mbase = maskw + (size_t)win * 2401;   // [a49*49 + j49]
  const float* rbase = rpb + hh;                     // [rel*8]

  for (int tile = wave; tile < 10; tile += 8) {
    const __bf16* Br = &sk[tile * 16 + l16][0];
    v16bf bk = frag_ld(Br + half * 8, Br + 16 + half * 8);
    v8f acc = {};
    acc = WMMA_BF16(aq, bk, acc);
    int j = tile * 16 + l16;
    bool jv = j < 147;
    int j49 = jv ? j % 49 : 0;
    int bh = j49 / 7, bw = j49 % 7;
    #pragma unroll
    for (int rr = 0; rr < 8; ++rr) {
      float v;
      if (jv) {
        int rel = (ahr[rr] - bh + 6) * 13 + (awr[rr] - bw + 6);
        v = acc[rr] * scale + rbase[rel * 8] + mbase[a49r[rr] * 49 + j49];
      } else {
        v = -100000.0f;
      }
      ss[half * 8 + rr][j] = v;
    }
  }
  __syncthreads();

  // parallel softmax: 16 threads per row, stride-16 scans + LDS reduction
  {
    int row = tid >> 4, sub = tid & 15;
    float mx = -1e30f;
    for (int j = sub; j < 147; j += 16) mx = fmaxf(mx, ss[row][j]);
    sred[row][sub] = mx;
    __syncthreads();
    mx = sred[row][0];
    #pragma unroll
    for (int i = 1; i < 16; ++i) mx = fmaxf(mx, sred[row][i]);
    float sum = 0.0f;
    for (int j = sub; j < 147; j += 16) {
      float e = __expf(ss[row][j] - mx);  ss[row][j] = e;  sum += e;
    }
    __syncthreads();
    sred[row][sub] = sum;
    __syncthreads();
    sum = 0.0f;
    #pragma unroll
    for (int i = 0; i < 16; ++i) sum += sred[row][i];
    float inv = 1.0f / sum;
    for (int j = sub; j < 160; j += 16)
      sp[row][j] = (j < 147) ? (__bf16)(ss[row][j] * inv) : (__bf16)0.0f;
  }
  __syncthreads();

  if (wave < 2) {                 // out 16x32 = P(16x160) @ V(160x32)
    v8f acc = {};
    for (int kk = 0; kk < 5; ++kk) {
      const __bf16* Ar = &sp[l16][kk * 32];
      v16bf ap = frag_ld(Ar + half * 8, Ar + 16 + half * 8);
      int n = wave * 16 + l16;
      v16bf bv;
      #pragma unroll
      for (int i = 0; i < 16; ++i) bv[i] = sv[kk * 32 + half * 8 + i + (i >= 8 ? 8 : 0)][n];
      acc = WMMA_BF16(ap, bv, acc);
    }
    #pragma unroll
    for (int rr = 0; rr < 8; ++rr) {
      int nq = mt * 16 + half * 8 + rr;
      if (nq < 147) {
        int d = wave * 16 + l16;
        AO[((size_t)b_ * 147 + nq) * 256 + hh * 32 + d] = (__bf16)acc[rr];
      }
    }
  }
}

// ---------------------------------------------------------------------------
// post: window-reverse + roll(+3,+3) + residual -> T (f32) ; LN2 -> X2 (bf16)
// ---------------------------------------------------------------------------
__global__ void __launch_bounds__(256)
post_kernel(const float* __restrict__ P, const float* __restrict__ rgb,
            const float* __restrict__ hs, const float* __restrict__ dem,
            const float* __restrict__ g, const float* __restrict__ bvec,
            float* __restrict__ T, __bf16* __restrict__ X2) {
  __shared__ float red[8];
  int r = blockIdx.x;
  int c = threadIdx.x;
  int m = r / (8 * 3136);
  int rem = r % (8 * 3136);
  int b = rem / 3136, pos = rem % 3136;
  int h = pos / 56, w = pos % 56;
  int hu = (h + 53) % 56, wu = (w + 53) % 56;
  int wy = hu / 7, wh = hu % 7, wx = wu / 7, ww = wu % 7;
  int win = wy * 8 + wx, n = wh * 7 + ww;
  size_t prow = (size_t)(b * 64 + win) * 147 + m * 49 + n;
  const float* sc = (m == 0 ? rgb : (m == 1 ? hs : dem)) +
                    ((size_t)b * 3136 + pos) * 256;
  float t = sc[c] + P[prow * 256 + c];
  T[(size_t)r * 256 + c] = t;
  float mean = blk_sum256(t, red) * (1.0f / 256.0f);
  float d = t - mean;
  float var = blk_sum256(d * d, red) * (1.0f / 256.0f);
  float xh = d * rsqrtf(var + 1e-5f) * g[c] + bvec[c];
  X2[(size_t)r * 256 + c] = (__bf16)xh;
}

// ---------------------------------------------------------------------------
extern "C" void kernel_launch(void* const* d_in, const int* in_sizes, int n_in,
                              void* d_out, int out_size, void* d_ws, size_t ws_size,
                              hipStream_t stream) {
  (void)in_sizes; (void)n_in; (void)out_size; (void)ws_size;
  const float* rgb    = (const float*)d_in[0];
  const float* hs     = (const float*)d_in[1];
  const float* dem    = (const float*)d_in[2];
  const float* maskw  = (const float*)d_in[3];
  const float* n1g    = (const float*)d_in[4];
  const float* n1b    = (const float*)d_in[5];
  const float* qkv_w  = (const float*)d_in[6];
  const float* qkv_b  = (const float*)d_in[7];
  const float* rpb    = (const float*)d_in[8];
  const float* proj_w = (const float*)d_in[9];
  const float* proj_b = (const float*)d_in[10];
  const float* n2g    = (const float*)d_in[11];
  const float* n2b    = (const float*)d_in[12];
  const float* fc1_w  = (const float*)d_in[13];
  const float* fc1_b  = (const float*)d_in[14];
  const float* fc2_w  = (const float*)d_in[15];
  const float* fc2_b  = (const float*)d_in[16];
  float* out = (float*)d_out;

  char* ws = (char*)d_ws;
  const size_t ROWS = 75264;                      // 512*147 == 3*8*3136
  size_t off_R1 = 0;                              // 38.5 MB: X / AO / X2
  size_t off_R0 = ROWS * 256 * 2;                 // 154 MB: QKV / P / H
  size_t off_T  = off_R0 + ROWS * 1024 * 2;       // 77 MB: T
  size_t off_W  = off_T + ROWS * 256 * 4;         // bf16 weights

  __bf16* X    = (__bf16*)(ws + off_R1);
  __bf16* QKV  = (__bf16*)(ws + off_R0);
  __bf16* AO   = (__bf16*)(ws + off_R1);
  float*  Pbuf = (float*) (ws + off_R0);
  __bf16* X2   = (__bf16*)(ws + off_R1);
  __bf16* HB   = (__bf16*)(ws + off_R0);
  float*  T    = (float*) (ws + off_T);
  __bf16* Wqkv  = (__bf16*)(ws + off_W);
  __bf16* Wproj = (__bf16*)(ws + off_W + 393216);
  __bf16* Wfc1  = (__bf16*)(ws + off_W + 524288);
  __bf16* Wfc2  = (__bf16*)(ws + off_W + 1048576);

  wconv_kernel<<<(768 * 256 + 255) / 256, 256, 0, stream>>>(qkv_w, Wqkv, 768 * 256);
  wconv_kernel<<<(256 * 256 + 255) / 256, 256, 0, stream>>>(proj_w, Wproj, 256 * 256);
  wconv_kernel<<<(1024 * 256 + 255) / 256, 256, 0, stream>>>(fc1_w, Wfc1, 1024 * 256);
  wconv_kernel<<<(256 * 1024 + 255) / 256, 256, 0, stream>>>(fc2_w, Wfc2, 256 * 1024);

  prep_kernel<<<(int)ROWS, 256, 0, stream>>>(rgb, hs, dem, n1g, n1b, X);

  gemm_kernel<0><<<dim3(588, 12), 256, 0, stream>>>(X, Wqkv, qkv_b, QKV, nullptr,
                                                    75264, 768, 256);
  attn_kernel<<<dim3(512, 8, 10), 256, 0, stream>>>(QKV, rpb, maskw, AO);

  gemm_kernel<2><<<dim3(588, 4), 256, 0, stream>>>(AO, Wproj, proj_b, Pbuf, nullptr,
                                                   75264, 256, 256);
  post_kernel<<<(int)ROWS, 256, 0, stream>>>(Pbuf, rgb, hs, dem, n2g, n2b, T, X2);

  gemm_kernel<1><<<dim3(588, 16), 256, 0, stream>>>(X2, Wfc1, fc1_b, HB, nullptr,
                                                    75264, 1024, 256);
  gemm_kernel<3><<<dim3(588, 4), 256, 0, stream>>>(HB, Wfc2, fc2_b, out, T,
                                                   75264, 256, 1024);
}
